// CrossModalAttentionScorer_18408229831185
// MI455X (gfx1250) — matrix operand order repaired and checked by reading, not compiled, not run on hardware
//
#include <hip/hip_runtime.h>
#include <hip/hip_bf16.h>

typedef __attribute__((ext_vector_type(16))) _Float16 v16h;
typedef __attribute__((ext_vector_type(8)))  _Float16 v8h;
typedef __attribute__((ext_vector_type(8)))  float    v8f;

#define Bdim 64
#define Adim 256
#define Tdim 128
#define Ddim 512
#define Dqdim 768
#define Hdim 512
#define NEGV -1.0e9f

// ---------------- fragment load helpers (wave32 WMMA layouts) ----------------

// A-matrix fragment 16x32 f16 from row-major [.., ld] tile at origin `tile`
// lane L: m = L%16 ; elements 0..15 -> k = (L/16)*16 + 0..15
__device__ __forceinline__ v16h load_a_frag(const _Float16* tile, int ld, int lane) {
    int h = lane >> 4, mr = lane & 15;
    const _Float16* p = tile + (size_t)mr * ld + h * 16;
    v8h lo = *(const v8h*)(p);
    v8h hi = *(const v8h*)(p + 8);
    return __builtin_shufflevector(lo, hi, 0,1,2,3,4,5,6,7,8,9,10,11,12,13,14,15);
}

// B-matrix fragment 32x16 f16 from n-major (transposed) tile Bt[n][k], origin `tileT`
// lane L: n = L%16 ; k = (L/16)*16 + e  -> one contiguous 32B load
__device__ __forceinline__ v16h load_b_frag(const _Float16* tileT, int ld, int lane) {
    int nr = lane & 15, kh = (lane >> 4) * 16;
    return *(const v16h*)(tileT + (size_t)nr * ld + kh);
}

// ---------------- CDNA5 async global->LDS copy (ASYNCcnt) ----------------

// copies 16 bytes per lane: global gaddr -> LDS lds_byte_addr
__device__ __forceinline__ void async_g2l_b128(unsigned lds_byte_addr, const void* gaddr) {
    asm volatile("global_load_async_to_lds_b128 %0, %1, off"
                 :: "v"(lds_byte_addr), "v"(gaddr)
                 : "memory");
}
__device__ __forceinline__ void wait_async0() {
    asm volatile("s_wait_asynccnt 0" ::: "memory");
}

// ---------------- elementwise / prep kernels ----------------

// W[k*N+n] (f32) -> Wt[n*K+k] (f16)
__global__ void conv_transpose_w(const float* __restrict__ W, _Float16* __restrict__ Wt,
                                 int K, int N) {
    int idx = blockIdx.x * blockDim.x + threadIdx.x;
    if (idx < K * N) {
        int k = idx / N, n = idx % N;
        Wt[(size_t)n * K + k] = (_Float16)W[idx];
    }
}

// per-row L2 normalize (f32 in, f16 out), one block per row
__global__ void l2norm_rows(const float* __restrict__ x, _Float16* __restrict__ out, int D) {
    __shared__ float red[8];
    int row = blockIdx.x;
    const float* p = x + (size_t)row * D;
    float s = 0.f;
    for (int i = threadIdx.x; i < D; i += blockDim.x) { float v = p[i]; s += v * v; }
    for (int m = 1; m < 32; m <<= 1) s += __shfl_xor(s, m, 32);
    int wv = threadIdx.x >> 5, lane = threadIdx.x & 31;
    if (lane == 0) red[wv] = s;
    __syncthreads();
    float tot = 0.f;
    for (int w = 0; w < (int)(blockDim.x >> 5); w++) tot += red[w];
    float inv = rsqrtf(fmaxf(tot, 1e-12f));
    _Float16* o = out + (size_t)row * D;
    for (int i = threadIdx.x; i < D; i += blockDim.x) o[i] = (_Float16)(p[i] * inv);
}

// combined = [anchor | attended | anchor*attended], f16 rows of 1536
__global__ void build_combined(const float* __restrict__ anchor,
                               const _Float16* __restrict__ att,
                               _Float16* __restrict__ comb) {
    int row = blockIdx.x;
    const float*    a = anchor + (size_t)row * Ddim;
    const _Float16* t = att    + (size_t)row * Hdim;
    _Float16*       c = comb   + (size_t)row * (Ddim + 2 * Hdim);
    for (int i = threadIdx.x; i < Ddim; i += blockDim.x) {
        float av = a[i];
        float tv = (float)t[i];
        c[i]              = (_Float16)av;
        c[Ddim + i]       = (_Float16)tv;
        c[2 * Ddim + i]   = (_Float16)(av * tv);
    }
}

// ---------------- pipelined WMMA GEMM: out[M x 512] = A[M x K] @ Bt^T ----------------
// Block: 128 threads (4 waves), tile M=32 x N=256; wave tile 32x64 (acc[2][4]).
// A tile (32x32 f16) staged in LDS via async copy, double buffered; B fragments
// register double-buffered so loads overlap the 8 WMMAs of the previous step.
template<int K>
__global__ void wmma_gemm_async(const _Float16* __restrict__ A,
                                const _Float16* __restrict__ Bt,
                                const float* __restrict__ bias,
                                _Float16* __restrict__ outH,
                                float* __restrict__ outF,
                                _Float16* __restrict__ outT, int rowsPerBatch,
                                int relu) {
    __shared__ __align__(16) _Float16 sA[2][32 * 32];
    int tid  = threadIdx.x;
    int lane = tid & 31;
    int wv   = tid >> 5;
    int m0 = (blockIdx.x >> 1) * 32;
    int n0 = (blockIdx.x & 1) * 256 + wv * 64;
    int h = lane >> 4, nr = lane & 15;

    // per-thread async-copy slot: 128 threads x 16B = 32x32 f16 tile
    int arow   = tid >> 2;        // 0..31
    int achunk = (tid & 3) * 8;   // half-element offset within row
    const _Float16* gA = A + (size_t)(m0 + arow) * K + achunk;
    unsigned lds0 = (unsigned)(uintptr_t)&sA[0][arow * 32 + achunk];
    unsigned lds1 = (unsigned)(uintptr_t)&sA[1][arow * 32 + achunk];

    // per-lane B base: n = n0 + nr, k-half select
    const _Float16* pB = Bt + (size_t)(n0 + nr) * K + h * 16;

    v8f zero = {};
    v8f acc[2][4];
#pragma unroll
    for (int mt = 0; mt < 2; mt++)
#pragma unroll
        for (int nt = 0; nt < 4; nt++) acc[mt][nt] = zero;

    v16h bc[4], bn[4];

    // prologue: stage k=0
    async_g2l_b128(lds0, gA);
#pragma unroll
    for (int nt = 0; nt < 4; nt++) bc[nt] = *(const v16h*)(pB + (size_t)nt * 16 * K);
    wait_async0();
    __syncthreads();

    const int NK = K / 32;
    int p = 0;
    for (int i = 0; i < NK; i++) {
        if (i + 1 < NK) {
            // prefetch next k-step: A tile async -> other LDS buffer, B frags -> regs
            async_g2l_b128(p ? lds0 : lds1, gA + (i + 1) * 32);
            const _Float16* pBn = pB + (size_t)(i + 1) * 32;
#pragma unroll
            for (int nt = 0; nt < 4; nt++) bn[nt] = *(const v16h*)(pBn + (size_t)nt * 16 * K);
        }
        const _Float16* s = &sA[p][0];
        v16h af0 = load_a_frag(s, 32, lane);            // rows m0..m0+15
        v16h af1 = load_a_frag(s + 16 * 32, 32, lane);  // rows m0+16..m0+31
#pragma unroll
        for (int nt = 0; nt < 4; nt++) {
            acc[0][nt] = __builtin_amdgcn_wmma_f32_16x16x32_f16(
                false, af0, false, bc[nt], (short)0, acc[0][nt], false, false);
            acc[1][nt] = __builtin_amdgcn_wmma_f32_16x16x32_f16(
                false, af1, false, bc[nt], (short)0, acc[1][nt], false, false);
        }
        wait_async0();
        __syncthreads();
#pragma unroll
        for (int nt = 0; nt < 4; nt++) bc[nt] = bn[nt];
        p ^= 1;
    }

#pragma unroll
    for (int mt = 0; mt < 2; mt++)
#pragma unroll
        for (int nt = 0; nt < 4; nt++) {
            int n = n0 + nt * 16 + nr;
            float bv = bias ? bias[n] : 0.f;
#pragma unroll
            for (int r = 0; r < 8; r++) {
                int m = m0 + mt * 16 + h * 8 + r;
                float v = acc[mt][nt][r] + bv;
                if (relu) v = fmaxf(v, 0.f);
                if (outH) outH[(size_t)m * 512 + n] = (_Float16)v;
                if (outF) outF[(size_t)m * 512 + n] = v;
                if (outT) {
                    int bb = m / rowsPerBatch, rr = m % rowsPerBatch;
                    outT[((size_t)(bb * 512 + n)) * rowsPerBatch + rr] = (_Float16)v;
                }
            }
        }
}

// ---------------- fused attention: scores -> mask -> softmax -> attended ----------------
// wave tile: (b, 16 anchor rows) x all T=128.
__global__ void attention_fused(const _Float16* __restrict__ Rp,   // [B*A, H] f16
                                const _Float16* __restrict__ Qp,   // [B, T, H] f16
                                const _Float16* __restrict__ QpT,  // [B, H, T] f16
                                const int* __restrict__ qmask,     // [B, T]
                                float* __restrict__ scores_out,    // [B, A, T] f32
                                _Float16* __restrict__ att_out) {  // [B*A, H] f16
    __shared__ _Float16 sAttn[4][16 * 128];
    int lane = threadIdx.x & 31;
    int wv   = threadIdx.x >> 5;
    int tileId = blockIdx.x * 4 + wv;          // 0..1023
    int b  = tileId >> 4;                       // 16 a-tiles per batch
    int a0 = (tileId & 15) * 16;
    int h = lane >> 4, nr = lane & 15;

    const _Float16* Arow = Rp + (size_t)(b * Adim + a0) * Hdim;
    const _Float16* Bq   = Qp + (size_t)b * Tdim * Hdim;

    v8f zero = {};
    v8f acc[8];
#pragma unroll
    for (int i = 0; i < 8; i++) acc[i] = zero;

    // scores[16 x 128] = Rp_tile[16 x 512] @ Qp[b]^T
    // all 8 B-frags live simultaneously so their loads overlap the WMMAs
    for (int k0 = 0; k0 < Hdim; k0 += 32) {
        v16h af = load_a_frag(Arow + k0, Hdim, lane);
        v16h bf[8];
#pragma unroll
        for (int nt = 0; nt < 8; nt++)
            bf[nt] = load_b_frag(Bq + (size_t)(nt * 16) * Hdim + k0, Hdim, lane);
#pragma unroll
        for (int nt = 0; nt < 8; nt++)
            acc[nt] = __builtin_amdgcn_wmma_f32_16x16x32_f16(
                false, af, false, bf[nt], (short)0, acc[nt], false, false);
    }

    // mask + write raw scores
#pragma unroll
    for (int nt = 0; nt < 8; nt++) {
        int t = nt * 16 + nr;
        bool valid = qmask[b * Tdim + t] > 0;
#pragma unroll
        for (int r = 0; r < 8; r++) {
            float v = valid ? acc[nt][r] : NEGV;
            acc[nt][r] = v;
            int arow = a0 + h * 8 + r;
            scores_out[((size_t)(b * Adim + arow)) * Tdim + t] = v;
        }
    }

    // softmax over T per row (row index = h*8 + r); reduce across 16-lane N-group
    float mx[8], sm[8];
#pragma unroll
    for (int r = 0; r < 8; r++) mx[r] = -3.4e38f;
#pragma unroll
    for (int nt = 0; nt < 8; nt++)
#pragma unroll
        for (int r = 0; r < 8; r++) mx[r] = fmaxf(mx[r], acc[nt][r]);
#pragma unroll
    for (int r = 0; r < 8; r++)
        for (int s = 1; s < 16; s <<= 1) mx[r] = fmaxf(mx[r], __shfl_xor(mx[r], s, 32));
#pragma unroll
    for (int r = 0; r < 8; r++) sm[r] = 0.f;
#pragma unroll
    for (int nt = 0; nt < 8; nt++)
#pragma unroll
        for (int r = 0; r < 8; r++) {
            float p = __expf(acc[nt][r] - mx[r]);
            acc[nt][r] = p;
            sm[r] += p;
        }
#pragma unroll
    for (int r = 0; r < 8; r++)
        for (int s = 1; s < 16; s <<= 1) sm[r] += __shfl_xor(sm[r], s, 32);

    // write attn (f16) to LDS in row-major [16 x 128] for A-fragment reload
    _Float16* my = sAttn[wv];
#pragma unroll
    for (int nt = 0; nt < 8; nt++) {
        int t = nt * 16 + nr;
#pragma unroll
        for (int r = 0; r < 8; r++)
            my[(h * 8 + r) * 128 + t] = (_Float16)(acc[nt][r] / sm[r]);
    }
    __syncthreads();

    // attended[16 x 512] = attn[16 x 128] @ Qp[b][128 x 512]
    v16h pf[4];
#pragma unroll
    for (int kk = 0; kk < 4; kk++) pf[kk] = load_a_frag(my + kk * 32, 128, lane);

    const _Float16* Bt0 = QpT + (size_t)b * Hdim * Tdim;
    v16h bcur[4], bnxt[4];
#pragma unroll
    for (int kk = 0; kk < 4; kk++)
        bcur[kk] = load_b_frag(Bt0 + kk * 32, Tdim, lane);

    for (int n0 = 0; n0 < Hdim; n0 += 16) {
        if (n0 + 16 < Hdim) {
#pragma unroll
            for (int kk = 0; kk < 4; kk++)
                bnxt[kk] = load_b_frag(Bt0 + (size_t)(n0 + 16) * Tdim + kk * 32, Tdim, lane);
        }
        v8f acc2 = zero;
#pragma unroll
        for (int kk = 0; kk < 4; kk++)
            acc2 = __builtin_amdgcn_wmma_f32_16x16x32_f16(
                false, pf[kk], false, bcur[kk], (short)0, acc2, false, false);
#pragma unroll
        for (int r = 0; r < 8; r++) {
            int m = a0 + h * 8 + r;
            att_out[((size_t)(b * Adim + m)) * Hdim + (n0 + nr)] = (_Float16)acc2[r];
        }
#pragma unroll
        for (int kk = 0; kk < 4; kk++) bcur[kk] = bnxt[kk];
    }
}

// ---------------- host launch ----------------

extern "C" void kernel_launch(void* const* d_in, const int* in_sizes, int n_in,
                              void* d_out, int out_size, void* d_ws, size_t ws_size,
                              hipStream_t stream) {
    const float* anchor = (const float*)d_in[0];   // [B,A,D]
    const float* query  = (const float*)d_in[1];   // [B,T,Dq]
    const int*   qmask  = (const int*)d_in[2];     // [B,T]
    const float* Wr     = (const float*)d_in[3];   // [D,H]
    const float* Wq     = (const float*)d_in[4];   // [Dq,H]
    const float* W1     = (const float*)d_in[5];   // [D+2H,H]
    const float* b1     = (const float*)d_in[6];   // [H]
    const float* W2     = (const float*)d_in[7];   // [H,H]
    const float* b2     = (const float*)d_in[8];   // [H]

    float* out_x      = (float*)d_out;                               // [B,A,H]
    float* out_scores = out_x + (size_t)Bdim * Adim * Hdim;          // [B,A,T]

    // workspace carve-up (all f16)
    char* w = (char*)d_ws;
    size_t off = 0;
    auto take = [&](size_t n_half) {
        void* p = w + off;
        off += ((n_half * 2 + 255) & ~(size_t)255);
        return p;
    };
    _Float16* Wrt  = (_Float16*)take((size_t)Hdim * Ddim);
    _Float16* Wqt  = (_Float16*)take((size_t)Hdim * Dqdim);
    _Float16* W1t  = (_Float16*)take((size_t)Hdim * (Ddim + 2 * Hdim));
    _Float16* W2t  = (_Float16*)take((size_t)Hdim * Hdim);
    _Float16* An   = (_Float16*)take((size_t)Bdim * Adim * Ddim);
    _Float16* Qn   = (_Float16*)take((size_t)Bdim * Tdim * Dqdim);
    _Float16* Rp   = (_Float16*)take((size_t)Bdim * Adim * Hdim);
    _Float16* Qp   = (_Float16*)take((size_t)Bdim * Tdim * Hdim);
    _Float16* QpT  = (_Float16*)take((size_t)Bdim * Hdim * Tdim);
    _Float16* Att  = (_Float16*)take((size_t)Bdim * Adim * Hdim);
    _Float16* Comb = (_Float16*)take((size_t)Bdim * Adim * (Ddim + 2 * Hdim));
    _Float16* X1   = (_Float16*)take((size_t)Bdim * Adim * Hdim);

    // 1) weights -> f16, transposed (n-major for B-fragment loads)
    conv_transpose_w<<<(Ddim * Hdim + 255) / 256, 256, 0, stream>>>(Wr, Wrt, Ddim, Hdim);
    conv_transpose_w<<<(Dqdim * Hdim + 255) / 256, 256, 0, stream>>>(Wq, Wqt, Dqdim, Hdim);
    conv_transpose_w<<<((Ddim + 2 * Hdim) * Hdim + 255) / 256, 256, 0, stream>>>(
        W1, W1t, Ddim + 2 * Hdim, Hdim);
    conv_transpose_w<<<(Hdim * Hdim + 255) / 256, 256, 0, stream>>>(W2, W2t, Hdim, Hdim);

    // 2) l2 normalize -> f16
    l2norm_rows<<<Bdim * Adim, 256, 0, stream>>>(anchor, An, Ddim);
    l2norm_rows<<<Bdim * Tdim, 256, 0, stream>>>(query, Qn, Dqdim);

    // 3) projections (pipelined WMMA GEMMs; grid = (M/32) * 2 n-halves)
    wmma_gemm_async<Ddim><<<(Bdim * Adim / 32) * 2, 128, 0, stream>>>(
        An, Wrt, nullptr, Rp, nullptr, nullptr, 1, 0);
    wmma_gemm_async<Dqdim><<<(Bdim * Tdim / 32) * 2, 128, 0, stream>>>(
        Qn, Wqt, nullptr, Qp, nullptr, QpT, Tdim, 0);

    // 4) fused attention (scores + mask + softmax + attended)
    attention_fused<<<(Bdim * Adim / 16) / 4, 128, 0, stream>>>(
        Rp, Qp, QpT, qmask, out_scores, Att);

    // 5) combined features
    build_combined<<<Bdim * Adim, 256, 0, stream>>>(anchor, Att, Comb);

    // 6) MLP
    wmma_gemm_async<Ddim + 2 * Hdim><<<(Bdim * Adim / 32) * 2, 128, 0, stream>>>(
        Comb, W1t, b1, X1, nullptr, nullptr, 1, 1);
    wmma_gemm_async<Hdim><<<(Bdim * Adim / 32) * 2, 128, 0, stream>>>(
        X1, W2t, b2, nullptr, out_x, nullptr, 1, 0);
}
